// RotaryMHA_60902636257496
// MI455X (gfx1250) — matrix operand order repaired and checked by reading, not compiled, not run on hardware
//
#include <hip/hip_runtime.h>
#include <hip/hip_bf16.h>

#define T_SEQ  2048
#define BATCH  4
#define DMODEL 1024
#define NHEAD  16
#define HD     64
#define NTOK   (T_SEQ * BATCH)   // 8192

typedef __attribute__((ext_vector_type(16))) __bf16 v16bf;
typedef __attribute__((ext_vector_type(8)))  float  v8f;

__device__ __forceinline__ v8f wmma_bf16(v16bf a, v16bf b, v8f c) {
    return __builtin_amdgcn_wmma_f32_16x16x32_bf16(
        false, a, false, b, (short)0, c, false, false);
}

// CDNA5 async global->LDS copy, 16 bytes per lane.
__device__ __forceinline__ void async_copy_b128(unsigned lds_off, const void* gptr) {
    asm volatile("global_load_async_to_lds_b128 %0, %1, off"
                 :: "v"(lds_off), "v"(gptr) : "memory");
}
__device__ __forceinline__ void wait_async0() {
    asm volatile("s_wait_asynccnt 0x0" ::: "memory");
}
__device__ __forceinline__ unsigned lds_off_of(const void* p) {
    return (unsigned)(uintptr_t)p;  // generic ptr to LDS: low 32 bits = LDS offset
}

// XOR-butterfly lane exchange via ds_swizzle (immediate pattern, no index VGPR)
template <int XORM>
__device__ __forceinline__ float swz_xor(float x) {
    return __int_as_float(
        __builtin_amdgcn_ds_swizzle(__float_as_int(x), (XORM << 10) | 0x1f));
}
__device__ __forceinline__ float red_max16(float x) {
    x = fmaxf(x, swz_xor<8>(x)); x = fmaxf(x, swz_xor<4>(x));
    x = fmaxf(x, swz_xor<2>(x)); x = fmaxf(x, swz_xor<1>(x));
    return x;
}
__device__ __forceinline__ float red_sum16(float x) {
    x += swz_xor<8>(x); x += swz_xor<4>(x);
    x += swz_xor<2>(x); x += swz_xor<1>(x);
    return x;
}

// A fragment 16x32 (MxK), row-major [m][k], leading dim ld (halves).
__device__ __forceinline__ v16bf load_a_frag(const __bf16* base, int row0, int ld, int kb) {
    int lane = threadIdx.x & 31;
    const __bf16* p = base + (size_t)(row0 + (lane & 15)) * ld + kb;
    int off = (lane & 16) ? 16 : 0;
    v16bf f;
#pragma unroll
    for (int i = 0; i < 8; ++i) f[i] = p[off + i];
#pragma unroll
    for (int i = 0; i < 8; ++i) f[8 + i] = p[off + 8 + i];
    return f;
}

// B fragment 32x16 (KxN), "n-major" storage: base[n*ld + k].
__device__ __forceinline__ v16bf load_b_frag(const __bf16* base, int n0, int ld, int kb) {
    int lane = threadIdx.x & 31;
    const __bf16* p = base + (size_t)(n0 + (lane & 15)) * ld + kb + ((lane & 16) ? 16 : 0);
    v16bf f;
#pragma unroll
    for (int i = 0; i < 16; ++i) f[i] = p[i];
    return f;
}

// ---------------------------------------------------------------------------
__global__ void cvt_bf16_kernel(const float* __restrict__ in, __bf16* __restrict__ out, int n) {
    int i = blockIdx.x * blockDim.x + threadIdx.x;
    if (i < n) out[i] = (__bf16)in[i];
}

// ---------------------------------------------------------------------------
// WMMA GEMM, double-buffered async tiles. Block 256 thr, tile M=128 N=64.
// MODE 0: A = x bf16 row-major; out bf16 scattered to [(b*H+h)*T + t][d]
// MODE 1: A gathered from bh-layout; out f32 row-major (d_out)
// ---------------------------------------------------------------------------
#define LDA 40   // 32 + 8 pad; row stride 80B (16B multiple)
template <int MODE>
__global__ void wmma_gemm_kernel(const __bf16* __restrict__ A,
                                 const __bf16* __restrict__ W,
                                 const float* __restrict__ bias,
                                 void* __restrict__ out) {
    __shared__ __bf16 Atile[2][128 * LDA];
    __shared__ __bf16 Btile[2][64 * LDA];

    const int rbase = blockIdx.x * 128;
    const int n0    = blockIdx.y * 64;
    const int w     = threadIdx.x >> 5;
    const int lane  = threadIdx.x & 31;
    const int wm    = w & 3;
    const int wn    = w >> 2;
    const unsigned at_off = lds_off_of(Atile);
    const unsigned bt_off = lds_off_of(Btile);

    auto issue_tiles = [&](int k0, int buf) {
        unsigned abase = at_off + (unsigned)buf * (128 * LDA * 2);
        for (int c = threadIdx.x; c < 512; c += 256) {
            int row = c >> 2, seg = (c & 3) * 8;
            int r = rbase + row, kk = k0 + seg;
            const __bf16* src;
            if (MODE == 0) {
                src = A + (size_t)r * DMODEL + kk;
            } else {
                int t = r >> 2, b = r & 3, h = kk >> 6, d = kk & 63;
                src = A + ((size_t)(b * NHEAD + h) * T_SEQ + t) * HD + d;
            }
            async_copy_b128(abase + (unsigned)(row * LDA + seg) * 2, src);
        }
        unsigned bbase = bt_off + (unsigned)buf * (64 * LDA * 2);
        int c = threadIdx.x;
        int n = c >> 2, seg = (c & 3) * 8;
        async_copy_b128(bbase + (unsigned)(n * LDA + seg) * 2,
                        W + (size_t)(n0 + n) * DMODEL + k0 + seg);
    };

    v8f acc[2][2] = {};

    issue_tiles(0, 0);
    wait_async0();
    __syncthreads();

    int cur = 0;
    for (int k0 = 0; k0 < DMODEL; k0 += 32) {
        if (k0 + 32 < DMODEL) issue_tiles(k0 + 32, cur ^ 1);  // overlap with compute

        const __bf16* At = &Atile[cur][0];
        const __bf16* Bt = &Btile[cur][0];
        v16bf a0 = load_a_frag(At, wm * 32,      LDA, 0);
        v16bf a1 = load_a_frag(At, wm * 32 + 16, LDA, 0);
        v16bf b0 = load_b_frag(Bt, wn * 32,      LDA, 0);
        v16bf b1 = load_b_frag(Bt, wn * 32 + 16, LDA, 0);

        acc[0][0] = wmma_bf16(a0, b0, acc[0][0]);
        acc[0][1] = wmma_bf16(a0, b1, acc[0][1]);
        acc[1][0] = wmma_bf16(a1, b0, acc[1][0]);
        acc[1][1] = wmma_bf16(a1, b1, acc[1][1]);

        wait_async0();
        __syncthreads();
        cur ^= 1;
    }

    const int colloc = lane & 15;
    const int rowoff = (lane & 16) ? 8 : 0;
#pragma unroll
    for (int i = 0; i < 2; ++i)
#pragma unroll
        for (int j = 0; j < 2; ++j) {
            int ng = n0 + wn * 32 + j * 16 + colloc;
            float bv = bias[ng];
#pragma unroll
            for (int vg = 0; vg < 8; ++vg) {
                int rg = rbase + wm * 32 + i * 16 + rowoff + vg;
                float val = acc[i][j][vg] + bv;
                if (MODE == 0) {
                    int t = rg >> 2, b = rg & 3, h = ng >> 6, d = ng & 63;
                    ((__bf16*)out)[((size_t)(b * NHEAD + h) * T_SEQ + t) * HD + d] = (__bf16)val;
                } else {
                    ((float*)out)[(size_t)rg * DMODEL + ng] = val;
                }
            }
        }
}

// ---------------------------------------------------------------------------
// RoPE in place on bf16 [bh][t][d]; one thread per (row, j<32) pair
// ---------------------------------------------------------------------------
__global__ void rope_kernel(__bf16* __restrict__ x) {
    int idx = blockIdx.x * blockDim.x + threadIdx.x;
    int j   = idx & 31;
    int row = idx >> 5;
    int t   = row & (T_SEQ - 1);
    float inv = __expf(-(float)j * (9.210340371976184f / 32.0f));  // 10000^{-j/32}
    float ang = (float)t * inv;
    float s, c;
    __sincosf(ang, &s, &c);
    __bf16* p = x + (size_t)row * HD;
    float x1 = (float)p[j], x2 = (float)p[j + 32];
    p[j]      = (__bf16)(c * x1 - s * x2);
    p[j + 32] = (__bf16)(c * x2 + s * x1);
}

// ---------------------------------------------------------------------------
// Flash attention, double-buffered K/V tiles. Grid (T/128, B*H).
// Block 256 = 8 waves, wave = 16 q rows. K async->LDS; V transposed via regs.
// ---------------------------------------------------------------------------
#define LDT 72   // 64 + 8 pad; row stride 144B (16B multiple)
__global__ void flash_attn_kernel(const __bf16* __restrict__ q,
                                  const __bf16* __restrict__ k,
                                  const __bf16* __restrict__ v,
                                  __bf16* __restrict__ o) {
    __shared__ __bf16 Kt[2][64 * LDT];
    __shared__ __bf16 Vt[2][64 * LDT];   // transposed: Vt[buf][d][key]
    __shared__ __bf16 Qt[8][16 * LDT];
    __shared__ __bf16 Pt[8][16 * LDT];

    const int bh    = blockIdx.y;
    const int w     = threadIdx.x >> 5;
    const int lane  = threadIdx.x & 31;
    const int qrow0 = blockIdx.x * 128 + w * 16;

    const __bf16* qbh = q + (size_t)bh * T_SEQ * HD;
    const __bf16* kbh = k + (size_t)bh * T_SEQ * HD;
    const __bf16* vbh = v + (size_t)bh * T_SEQ * HD;
    const unsigned kt_off = lds_off_of(Kt);

    // per-thread V chunk coordinates (2 chunks: c and c+256)
    const int vrow0 = threadIdx.x >> 3;            // 0..31   (key)
    const int vseg0 = (threadIdx.x & 7) * 8;       // d base
    const int vrow1 = vrow0 + 32;                  // 32..63

    auto issue_k = [&](int s0, int buf) {
        unsigned base = kt_off + (unsigned)buf * (64 * LDT * 2);
        for (int c = threadIdx.x; c < 512; c += 256) {
            int row = c >> 3, seg = (c & 7) * 8;
            async_copy_b128(base + (unsigned)(row * LDT + seg) * 2,
                            kbh + (size_t)(s0 + row) * HD + seg);
        }
    };
    auto load_v_regs = [&](int s0, uint4& r0, uint4& r1) {
        r0 = *(const uint4*)(vbh + (size_t)(s0 + vrow0) * HD + vseg0);
        r1 = *(const uint4*)(vbh + (size_t)(s0 + vrow1) * HD + vseg0);
    };
    auto store_v_lds = [&](int buf, uint4 r0, uint4 r1) {
        __bf16* dst = &Vt[buf][0];
        union { uint4 u; __bf16 h[8]; } cv;
        cv.u = r0;
#pragma unroll
        for (int i = 0; i < 8; ++i) dst[(vseg0 + i) * LDT + vrow0] = cv.h[i];
        cv.u = r1;
#pragma unroll
        for (int i = 0; i < 8; ++i) dst[(vseg0 + i) * LDT + vrow1] = cv.h[i];
    };

    // stage this wave's Q block (async) + preload tile 0
    {
        __bf16* qw = &Qt[w][0];
        unsigned qw_off = lds_off_of(qw);
        for (int c = lane; c < 128; c += 32) {
            int row = c >> 3, seg = (c & 7) * 8;
            async_copy_b128(qw_off + (unsigned)(row * LDT + seg) * 2,
                            qbh + (size_t)(qrow0 + row) * HD + seg);
        }
        issue_k(0, 0);
        uint4 r0, r1;
        load_v_regs(0, r0, r1);
        store_v_lds(0, r0, r1);
        wait_async0();
        __syncthreads();
    }

    v8f co[4] = {};
    float mrow[8], lrow[8];
#pragma unroll
    for (int i = 0; i < 8; ++i) { mrow[i] = -3.0e38f; lrow[i] = 0.0f; }

    int cur = 0;
    for (int s0 = 0; s0 < T_SEQ; s0 += 64) {
        const bool pf = (s0 + 64 < T_SEQ);
        uint4 vr0, vr1;
        if (pf) {                       // prefetch next tile while computing
            issue_k(s0 + 64, cur ^ 1);
            load_v_regs(s0 + 64, vr0, vr1);
        }

        // S = Q * K^T (16x64 per wave)
        v8f cs[4] = {};
        {
            const __bf16* qw = &Qt[w][0];
            const __bf16* Kc = &Kt[cur][0];
            v16bf aq0 = load_a_frag(qw, 0, LDT, 0);
            v16bf aq1 = load_a_frag(qw, 0, LDT, 32);
#pragma unroll
            for (int j = 0; j < 4; ++j) {
                cs[j] = wmma_bf16(aq0, load_b_frag(Kc, j * 16, LDT, 0),  cs[j]);
                cs[j] = wmma_bf16(aq1, load_b_frag(Kc, j * 16, LDT, 32), cs[j]);
            }
        }

        // online softmax; row = vg + 8*(lane>=16); 16 lanes of a half share a row
#pragma unroll
        for (int vg = 0; vg < 8; ++vg) {
            float mx = -3.0e38f;
#pragma unroll
            for (int j = 0; j < 4; ++j) {
                cs[j][vg] *= 0.125f;
                mx = fmaxf(mx, cs[j][vg]);
            }
            mx = red_max16(mx);
            float mnew  = fmaxf(mrow[vg], mx);
            float alpha = __expf(mrow[vg] - mnew);
            float rs = 0.0f;
#pragma unroll
            for (int j = 0; j < 4; ++j) {
                float e = __expf(cs[j][vg] - mnew);
                cs[j][vg] = e;
                rs += e;
            }
            rs = red_sum16(rs);
            lrow[vg] = lrow[vg] * alpha + rs;
            mrow[vg] = mnew;
#pragma unroll
            for (int j = 0; j < 4; ++j) co[j][vg] *= alpha;
        }

        // P -> LDS (A-fragment re-layout), then O += P * V
        __bf16* pw = &Pt[w][0];
        const int rowoff = (lane & 16) ? 8 : 0;
#pragma unroll
        for (int vg = 0; vg < 8; ++vg)
#pragma unroll
            for (int j = 0; j < 4; ++j)
                pw[(vg + rowoff) * LDT + j * 16 + (lane & 15)] = (__bf16)cs[j][vg];
        __syncthreads();

        {
            const __bf16* Vc = &Vt[cur][0];
            v16bf ap0 = load_a_frag(pw, 0, LDT, 0);
            v16bf ap1 = load_a_frag(pw, 0, LDT, 32);
#pragma unroll
            for (int j = 0; j < 4; ++j) {
                co[j] = wmma_bf16(ap0, load_b_frag(Vc, j * 16, LDT, 0),  co[j]);
                co[j] = wmma_bf16(ap1, load_b_frag(Vc, j * 16, LDT, 32), co[j]);
            }
        }

        if (pf) store_v_lds(cur ^ 1, vr0, vr1);  // other buffer: safe before barrier
        wait_async0();
        __syncthreads();
        cur ^= 1;
    }

    const int rowoff = (lane & 16) ? 8 : 0;
#pragma unroll
    for (int vg = 0; vg < 8; ++vg) {
        float inv = 1.0f / lrow[vg];
        int r = qrow0 + vg + rowoff;
#pragma unroll
        for (int j = 0; j < 4; ++j)
            o[((size_t)bh * T_SEQ + r) * HD + j * 16 + (lane & 15)] = (__bf16)(co[j][vg] * inv);
    }
}

// ---------------------------------------------------------------------------
extern "C" void kernel_launch(void* const* d_in, const int* in_sizes, int n_in,
                              void* d_out, int out_size, void* d_ws, size_t ws_size,
                              hipStream_t stream) {
    const float* x  = (const float*)d_in[0];
    const float* Wq = (const float*)d_in[1];
    const float* bq = (const float*)d_in[2];
    const float* Wk = (const float*)d_in[3];
    const float* bk = (const float*)d_in[4];
    const float* Wv = (const float*)d_in[5];
    const float* bv = (const float*)d_in[6];
    const float* Wo = (const float*)d_in[7];
    const float* bo = (const float*)d_in[8];

    char* ws = (char*)d_ws;
    const size_t SZ_X = (size_t)NTOK * DMODEL * 2;    // 16 MiB
    const size_t SZ_W = (size_t)DMODEL * DMODEL * 2;  //  2 MiB
    __bf16* xb  = (__bf16*)(ws);
    __bf16* wqb = (__bf16*)(ws + SZ_X);
    __bf16* wkb = (__bf16*)(ws + SZ_X + SZ_W);
    __bf16* wvb = (__bf16*)(ws + SZ_X + 2 * SZ_W);
    __bf16* wob = (__bf16*)(ws + SZ_X + 3 * SZ_W);
    __bf16* qb  = (__bf16*)(ws + SZ_X + 4 * SZ_W);
    __bf16* kb_ = (__bf16*)(ws + 2 * SZ_X + 4 * SZ_W);
    __bf16* vb_ = (__bf16*)(ws + 3 * SZ_X + 4 * SZ_W);
    __bf16* ab  = (__bf16*)(ws + 4 * SZ_X + 4 * SZ_W);

    const int nx = NTOK * DMODEL;
    const int nw = DMODEL * DMODEL;
    cvt_bf16_kernel<<<nx / 256, 256, 0, stream>>>(x,  xb,  nx);
    cvt_bf16_kernel<<<nw / 256, 256, 0, stream>>>(Wq, wqb, nw);
    cvt_bf16_kernel<<<nw / 256, 256, 0, stream>>>(Wk, wkb, nw);
    cvt_bf16_kernel<<<nw / 256, 256, 0, stream>>>(Wv, wvb, nw);
    cvt_bf16_kernel<<<nw / 256, 256, 0, stream>>>(Wo, wob, nw);

    dim3 ggrid(NTOK / 128, DMODEL / 64);
    wmma_gemm_kernel<0><<<ggrid, 256, 0, stream>>>(xb, wqb, bq, qb);
    wmma_gemm_kernel<0><<<ggrid, 256, 0, stream>>>(xb, wkb, bk, kb_);
    wmma_gemm_kernel<0><<<ggrid, 256, 0, stream>>>(xb, wvb, bv, vb_);

    const int nrope = BATCH * NHEAD * T_SEQ * 32;
    rope_kernel<<<nrope / 256, 256, 0, stream>>>(qb);
    rope_kernel<<<nrope / 256, 256, 0, stream>>>(kb_);

    flash_attn_kernel<<<dim3(T_SEQ / 128, BATCH * NHEAD), 256, 0, stream>>>(qb, kb_, vb_, ab);

    wmma_gemm_kernel<1><<<ggrid, 256, 0, stream>>>(ab, wob, bo, d_out);
}
